// L4maAttention_84404697301390
// MI455X (gfx1250) — matrix operand order repaired and checked by reading, not compiled, not run on hardware
//
#include <hip/hip_runtime.h>
#include <stdint.h>
#include <math.h>

#define NH   32
#define NKV  8
#define HD   128
#define SEQ  1024
#define BATCH 4
#define HID  4096
#define NNZ  (BATCH*SEQ)

typedef _Float16 half_t;
typedef __attribute__((ext_vector_type(16))) _Float16 v16h;
typedef __attribute__((ext_vector_type(8)))  _Float16 v8h;
typedef __attribute__((ext_vector_type(4)))  _Float16 v4h;
typedef __attribute__((ext_vector_type(8)))  float    v8f;
typedef __attribute__((ext_vector_type(4)))  uint32_t u32x4;
typedef __attribute__((ext_vector_type(4)))  int      i32x4;
typedef __attribute__((ext_vector_type(8)))  int      i32x8;

#ifndef __has_builtin
#define __has_builtin(x) 0
#endif
#if __has_builtin(__builtin_amdgcn_tensor_load_to_lds) && \
    __has_builtin(__builtin_amdgcn_s_wait_tensorcnt)
#define HAVE_TDM 1
#else
#define HAVE_TDM 0
#endif

__device__ __forceinline__ v8f wmma_f16(v16h a, v16h b, v8f c) {
  return __builtin_amdgcn_wmma_f32_16x16x32_f16(false, a, false, b,
                                                (short)0, c, false, false);
}

// ---------------------------------------------------------------------------
// Tensor Data Mover: 2D f16 tile load (global -> LDS) with HW LDS padding.
// D# per CDNA5 ISA (group0 128b + group1 256b; groups 2/3 zero => 2D).
//   pad_interval code: 3 -> every 16 dwords, 5 -> every 64 dwords
//   pad_amount  code: 3 -> insert 4 dwords (8 halfs)
// 6-arg toolchain variant: (g0, g1, g2, g3, g4, cpol).
// ---------------------------------------------------------------------------
__device__ __forceinline__ void tdm_load_tile_f16(
    uint32_t lds_off, const half_t* gaddr,
    uint32_t tile_d0, uint32_t tile_d1,       // elems per row, rows
    uint32_t stride0,                          // row stride in elems
    uint32_t pad_interval_code, uint32_t pad_amount_code) {
#if HAVE_TDM
  uint64_t ga = (uint64_t)(uintptr_t)gaddr;
  u32x4 g0;
  g0[0] = 1u;                                          // count=1 (valid user D#)
  g0[1] = lds_off;                                     // lds_addr [63:32]
  g0[2] = (uint32_t)(ga & 0xFFFFFFFFu);                // global_addr [95:64]
  g0[3] = (uint32_t)((ga >> 32) & 0x01FFFFFFu)         // global_addr [120:96]
        | (2u << 30);                                  // type=2 ("image")
  i32x8 g1;
  g1[0] = (int)((1u << 16)                             // data_size=1 (2B)
        | (1u << 20)                                   // pad_enable
        | (pad_interval_code << 22)
        | (pad_amount_code << 25));
  g1[1] = (int)((tile_d0 & 0xFFFFu) << 16);            // tensor_dim0[15:0]
  g1[2] = (int)(((tile_d0 >> 16) & 0xFFFFu)            // tensor_dim0[31:16]
        | ((tile_d1 & 0xFFFFu) << 16));                // tensor_dim1[15:0]
  g1[3] = (int)(((tile_d1 >> 16) & 0xFFFFu)            // tensor_dim1[31:16]
        | ((tile_d0 & 0xFFFFu) << 16));                // tile_dim0
  g1[4] = (int)(tile_d1 & 0xFFFFu);                    // tile_dim1 (tile_dim2=0)
  g1[5] = (int)stride0;                                // tensor_dim0_stride[31:0]
  g1[6] = 0;                                           // stride0 hi / stride1 lo
  g1[7] = 0;
  i32x4 z4 = {0, 0, 0, 0};
  i32x8 z8 = {0, 0, 0, 0, 0, 0, 0, 0};
  __builtin_amdgcn_tensor_load_to_lds(g0, g1, z4, z4, z8, 0);
#else
  (void)lds_off; (void)gaddr; (void)tile_d0; (void)tile_d1;
  (void)stride0; (void)pad_interval_code; (void)pad_amount_code;
#endif
}

__device__ __forceinline__ void tdm_wait() {
#if HAVE_TDM
  __builtin_amdgcn_s_wait_tensorcnt(0);
#endif
}

__device__ __forceinline__ uint32_t lds_addr_of(const void* p) {
  return (uint32_t)(uintptr_t)p;   // generic ptr to __shared__: low 32b = LDS offset
}

// ---------------------------------------------------------------------------
// GEMM: C[M,N] = A[M,K] * W[N,K]^T.  Block: 8 waves, tile 64(M) x 128(N).
// Each wave: 4 m-tiles vs one B fragment -> 4 WMMA per 16-dword weight load.
// A_F16 path: double-buffered TDM async tile loads (wave 0 drives the TDM).
// ---------------------------------------------------------------------------
template<bool A_F16, bool OUT_F32>
__global__ __launch_bounds__(256)
void gemm_wmma(const void* __restrict__ A_, const float* __restrict__ W,
               void* __restrict__ C_, int N, int K) {
  __shared__ __align__(16) half_t ldsA[2][64][40];  // 40-half stride (TDM-padded)

  const int tid  = threadIdx.x;
  const int lane = tid & 31;
  const int wave = tid >> 5;
  const int mBase = blockIdx.y * 64;
  const int n0 = blockIdx.x * 128 + wave * 16;
  const int l16 = lane & 15;
  const int hs  = lane >> 4;

  const v8f vzero = {0,0,0,0,0,0,0,0};
  v8f acc[4];
  #pragma unroll
  for (int mt = 0; mt < 4; ++mt) acc[mt] = vzero;

  const int sr = tid >> 2;             // staged row 0..63
  const int sc = (tid & 3) * 8;        // staged col (8 elems)

  constexpr bool TDM_PATH = (A_F16 && HAVE_TDM);
  if (TDM_PATH) {
    if (wave == 0)
      tdm_load_tile_f16(lds_addr_of(&ldsA[0][0][0]),
                        (const half_t*)A_ + (size_t)mBase * K,
                        32, 64, (uint32_t)K, 3, 3);
  }

  for (int k0 = 0; k0 < K; k0 += 32) {
    int cur = 0;
    if (TDM_PATH) {
      cur = (k0 >> 5) & 1;
      if (wave == 0) tdm_wait();                 // tile `cur` resident
      __syncthreads();
      if (wave == 0 && (k0 + 32) < K)            // prefetch next tile async
        tdm_load_tile_f16(lds_addr_of(&ldsA[cur ^ 1][0][0]),
                          (const half_t*)A_ + (size_t)mBase * K + k0 + 32,
                          32, 64, (uint32_t)K, 3, 3);
    } else {
      // manual cooperative stage (with f32->f16 convert if needed)
      if (A_F16) {
        const half_t* A = (const half_t*)A_;
        v8h av = *(const v8h*)(A + (size_t)(mBase + sr) * K + k0 + sc);
        *(v8h*)(&ldsA[0][sr][sc]) = av;
      } else {
        const float* A = (const float*)A_;
        const float4 a0 = *(const float4*)(A + (size_t)(mBase + sr) * K + k0 + sc);
        const float4 a1 = *(const float4*)(A + (size_t)(mBase + sr) * K + k0 + sc + 4);
        v8h hv;
        hv[0]=(half_t)a0.x; hv[1]=(half_t)a0.y; hv[2]=(half_t)a0.z; hv[3]=(half_t)a0.w;
        hv[4]=(half_t)a1.x; hv[5]=(half_t)a1.y; hv[6]=(half_t)a1.z; hv[7]=(half_t)a1.w;
        *(v8h*)(&ldsA[0][sr][sc]) = hv;
      }
      __syncthreads();
    }

    // --- B fragment: lane col = n0+l16, K floats hs*8..+8 and 16+hs*8..+8 from W row ---
    const float* wp = W + (size_t)(n0 + l16) * K + k0 + hs * 8;
    float4 w0 = ((const float4*)wp)[0];
    float4 w1 = ((const float4*)wp)[1];
    float4 w2 = ((const float4*)wp)[4];
    float4 w3 = ((const float4*)wp)[5];
    __builtin_prefetch(wp + 32, 0, 1);
    v16h b;
    b[0]=(half_t)w0.x;  b[1]=(half_t)w0.y;  b[2]=(half_t)w0.z;  b[3]=(half_t)w0.w;
    b[4]=(half_t)w1.x;  b[5]=(half_t)w1.y;  b[6]=(half_t)w1.z;  b[7]=(half_t)w1.w;
    b[8]=(half_t)w2.x;  b[9]=(half_t)w2.y;  b[10]=(half_t)w2.z; b[11]=(half_t)w2.w;
    b[12]=(half_t)w3.x; b[13]=(half_t)w3.y; b[14]=(half_t)w3.z; b[15]=(half_t)w3.w;

    // --- 4 A fragments (shared stage) x 1 B fragment -> 4 WMMA ---
    #pragma unroll
    for (int mt = 0; mt < 4; ++mt) {
      const half_t* arow = &ldsA[cur][mt * 16 + l16][0];
      v8h alo = *(const v8h*)(arow + hs * 8);
      v8h ahi = *(const v8h*)(arow + 16 + hs * 8);
      v16h a;
      #pragma unroll
      for (int i = 0; i < 8; ++i) { a[i] = alo[i]; a[8 + i] = ahi[i]; }
      acc[mt] = wmma_f16(a, b, acc[mt]);
    }

    if (!TDM_PATH) __syncthreads();
  }

  // --- store 4 D tiles ---
  const int col = n0 + l16;
  #pragma unroll
  for (int mt = 0; mt < 4; ++mt) {
    const int row0 = mBase + mt * 16 + hs * 8;
    if (OUT_F32) {
      float* C = (float*)C_;
      #pragma unroll
      for (int r = 0; r < 8; ++r) C[(size_t)(row0 + r) * N + col] = acc[mt][r];
    } else {
      half_t* C = (half_t*)C_;
      #pragma unroll
      for (int r = 0; r < 8; ++r) C[(size_t)(row0 + r) * N + col] = (half_t)acc[mt][r];
    }
  }
}

// ---------------------------------------------------------------------------
// Llama-3.1 RoPE, in-place on f16 activations [NNZ, heads, HD], rotate-half.
// ---------------------------------------------------------------------------
__global__ void rope_kernel(half_t* __restrict__ x,
                            const int* __restrict__ pos_ids, int heads) {
  int idx = blockIdx.x * blockDim.x + threadIdx.x;
  int total = NNZ * heads * (HD / 2);
  if (idx >= total) return;
  int i = idx & 63;
  int h = (idx >> 6) % heads;
  int t = idx / (heads * 64);

  float fr = __powf(500000.0f, -((float)(2 * i)) * (1.0f / 128.0f));
  float wavelen = 6.283185307179586f / fr;
  float f;
  if (wavelen < 2048.0f)      f = fr;
  else if (wavelen > 8192.0f) f = fr * 0.125f;
  else {
    float smooth = (8192.0f / wavelen - 1.0f) * (1.0f / 3.0f);
    f = (1.0f - smooth) * fr * 0.125f + smooth * fr;
  }
  float ang = (float)pos_ids[t] * f;
  float c = __cosf(ang), s = __sinf(ang);
  size_t base = (size_t)t * heads * HD + (size_t)h * HD;
  float x1 = (float)x[base + i], x2 = (float)x[base + i + 64];
  x[base + i]      = (half_t)(x1 * c - x2 * s);
  x[base + i + 64] = (half_t)(x2 * c + x1 * s);
}

// ---------------------------------------------------------------------------
// Flash attention, causal, GQA, all-WMMA.  K chunk staged by the TDM while
// threads stage transposed V; per-wave P re-layout via LDS with s_wait_dscnt.
// ---------------------------------------------------------------------------
__global__ __launch_bounds__(256)
void attn_kernel(const half_t* __restrict__ q, const half_t* __restrict__ k,
                 const half_t* __restrict__ v, half_t* __restrict__ out) {
  __shared__ __align__(16) half_t ldsK [32][136];   // keys x hd (TDM-padded)
  __shared__ __align__(16) half_t ldsVT[128][40];   // hd x keys (transposed)
  __shared__ __align__(16) half_t ldsP [8][16][40]; // per-wave P bounce

  const int tid  = threadIdx.x;
  const int lane = tid & 31;
  const int wave = tid >> 5;
  const int l16  = lane & 15;
  const int hs   = lane >> 4;

  const int bid = blockIdx.x;                 // B*NH*8 blocks
  const int qb  = bid & 7;
  const int h   = (bid >> 3) & 31;
  const int b   = bid >> 8;
  const int kvh = h >> 2;
  const int qr0 = qb * 128 + wave * 16;

  const float scale = 0.08838834764831845f;   // 1/sqrt(128), folded into Q

  // q fragments (A layout), pre-scaled
  v16h aq[4];
  {
    size_t qrow = ((size_t)(b * SEQ + qr0 + l16)) * (NH * HD) + (size_t)h * HD;
    #pragma unroll
    for (int kt = 0; kt < 4; ++kt) {
      int base = kt * 32 + hs * 8;
      v8h lo = *(const v8h*)(q + qrow + base);
      v8h hi = *(const v8h*)(q + qrow + base + 16);
      #pragma unroll
      for (int i = 0; i < 8; ++i) {
        aq[kt][i]     = (half_t)((float)lo[i] * scale);
        aq[kt][8 + i] = (half_t)((float)hi[i] * scale);
      }
    }
  }

  const v8f vzero = {0,0,0,0,0,0,0,0};
  v8f o[8];
  #pragma unroll
  for (int n = 0; n < 8; ++n) o[n] = vzero;
  float mrow[8], lrow[8];
  #pragma unroll
  for (int r = 0; r < 8; ++r) { mrow[r] = -3.0e38f; lrow[r] = 0.0f; }

  const int nchunk = qb * 4 + 4;
  const int skey = tid >> 3;
  const int sd0  = (tid & 7) * 16;

  for (int c = 0; c < nchunk; ++c) {
    const int kk0 = c * 32;
    size_t row = ((size_t)(b * SEQ + kk0 + skey)) * (NKV * HD)
               + (size_t)kvh * HD + sd0;
    // --- K chunk: async TDM tile load (overlaps manual V staging below) ---
    if (HAVE_TDM) {
      if (wave == 0)
        tdm_load_tile_f16(lds_addr_of(&ldsK[0][0]),
                          k + ((size_t)(b * SEQ + kk0)) * (NKV * HD)
                            + (size_t)kvh * HD,
                          HD, 32, NKV * HD, 5, 3);
    } else {
      v8h k0v = *(const v8h*)(k + row);
      v8h k1v = *(const v8h*)(k + row + 8);
      *(v8h*)(&ldsK[skey][sd0])     = k0v;
      *(v8h*)(&ldsK[skey][sd0 + 8]) = k1v;
    }
    // --- V chunk: manual transposed stage ---
    {
      v8h v0v = *(const v8h*)(v + row);
      v8h v1v = *(const v8h*)(v + row + 8);
      #pragma unroll
      for (int i = 0; i < 8; ++i) ldsVT[sd0 + i][skey]     = v0v[i];
      #pragma unroll
      for (int i = 0; i < 8; ++i) ldsVT[sd0 + 8 + i][skey] = v1v[i];
    }
    if (HAVE_TDM) { if (wave == 0) tdm_wait(); }
    __syncthreads();

    if (kk0 <= qr0 + 15) {
      // --- scores: two 16x16 tiles over 4 K-tiles ---
      v8f s0 = vzero, s1 = vzero;
      #pragma unroll
      for (int kt = 0; kt < 4; ++kt) {
        const half_t* kp0 = &ldsK[l16][kt * 32 + hs * 8];
        v8h b0l = *(const v8h*)kp0;
        v8h b0h = *(const v8h*)(kp0 + 16);
        v16h b0;
        #pragma unroll
        for (int i = 0; i < 8; ++i) { b0[i] = b0l[i]; b0[8 + i] = b0h[i]; }
        s0 = wmma_f16(aq[kt], b0, s0);

        const half_t* kp1 = &ldsK[16 + l16][kt * 32 + hs * 8];
        v8h b1l = *(const v8h*)kp1;
        v8h b1h = *(const v8h*)(kp1 + 16);
        v16h b1;
        #pragma unroll
        for (int i = 0; i < 8; ++i) { b1[i] = b1l[i]; b1[8 + i] = b1h[i]; }
        s1 = wmma_f16(aq[kt], b1, s1);
      }

      // --- causal mask + row max ---
      float rmax[8];
      bool v0ok[8], v1ok[8];
      #pragma unroll
      for (int r = 0; r < 8; ++r) {
        int qp  = qr0 + hs * 8 + r;
        int kp0 = kk0 + l16, kp1 = kp0 + 16;
        v0ok[r] = (kp0 <= qp); v1ok[r] = (kp1 <= qp);
        float sv0 = v0ok[r] ? s0[r] : -3.0e38f;
        float sv1 = v1ok[r] ? s1[r] : -3.0e38f;
        s0[r] = sv0; s1[r] = sv1;
        rmax[r] = fmaxf(sv0, sv1);
      }
      #pragma unroll
      for (int m = 1; m < 16; m <<= 1)
        #pragma unroll
        for (int r = 0; r < 8; ++r)
          rmax[r] = fmaxf(rmax[r], __shfl_xor(rmax[r], m, 32));

      // --- online softmax ---
      float corr[8], psum[8];
      #pragma unroll
      for (int r = 0; r < 8; ++r) {
        float mn = fmaxf(mrow[r], rmax[r]);
        corr[r] = __expf(mrow[r] - mn);
        mrow[r] = mn;
        float p0 = v0ok[r] ? __expf(s0[r] - mn) : 0.0f;
        float p1 = v1ok[r] ? __expf(s1[r] - mn) : 0.0f;
        s0[r] = p0; s1[r] = p1;
        psum[r] = p0 + p1;
      }
      #pragma unroll
      for (int m = 1; m < 16; m <<= 1)
        #pragma unroll
        for (int r = 0; r < 8; ++r)
          psum[r] += __shfl_xor(psum[r], m, 32);
      #pragma unroll
      for (int r = 0; r < 8; ++r) lrow[r] = lrow[r] * corr[r] + psum[r];
      #pragma unroll
      for (int n = 0; n < 8; ++n)
        #pragma unroll
        for (int r = 0; r < 8; ++r) o[n][r] *= corr[r];

      // --- P: D layout -> LDS -> A layout ---
      #pragma unroll
      for (int r = 0; r < 8; ++r) {
        int rl = hs * 8 + r;
        ldsP[wave][rl][l16]      = (half_t)s0[r];
        ldsP[wave][rl][16 + l16] = (half_t)s1[r];
      }
      asm volatile("s_wait_dscnt 0" ::: "memory");
      v16h ap;
      {
        const half_t* prow = &ldsP[wave][l16][0];
        v8h lo = *(const v8h*)(prow + hs * 8);
        v8h hi = *(const v8h*)(prow + 16 + hs * 8);
        #pragma unroll
        for (int i = 0; i < 8; ++i) { ap[i] = lo[i]; ap[8 + i] = hi[i]; }
      }

      // --- O += P(16x32) * V(32x16), 8 hd tiles ---
      #pragma unroll
      for (int n = 0; n < 8; ++n) {
        const half_t* vp = &ldsVT[n * 16 + l16][hs * 8];
        v8h lo = *(const v8h*)vp;
        v8h hi = *(const v8h*)(vp + 16);
        v16h bv;
        #pragma unroll
        for (int i = 0; i < 8; ++i) { bv[i] = lo[i]; bv[8 + i] = hi[i]; }
        o[n] = wmma_f16(ap, bv, o[n]);
      }
    }
    __syncthreads();
  }

  // --- epilogue ---
  #pragma unroll
  for (int n = 0; n < 8; ++n)
    #pragma unroll
    for (int r = 0; r < 8; ++r) {
      int row = qr0 + hs * 8 + r;
      float val = o[n][r] / lrow[r];
      out[((size_t)(b * SEQ + row)) * (NH * HD) + (size_t)h * HD + n * 16 + l16]
          = (half_t)val;
    }
}

// ---------------------------------------------------------------------------
extern "C" void kernel_launch(void* const* d_in, const int* in_sizes, int n_in,
                              void* d_out, int out_size, void* d_ws, size_t ws_size,
                              hipStream_t stream) {
  (void)in_sizes; (void)n_in; (void)out_size; (void)ws_size;
  const float* hidden = (const float*)d_in[0];
  const float* wq     = (const float*)d_in[1];
  const float* wk     = (const float*)d_in[2];
  const float* wv     = (const float*)d_in[3];
  const float* wo     = (const float*)d_in[4];
  const int*   pos    = (const int*)d_in[6];

  half_t* qf = (half_t*)d_ws;                          // [NNZ, NH*HD]  32 MB
  half_t* kf = qf + (size_t)NNZ * NH * HD;             // [NNZ, NKV*HD]  8 MB
  half_t* vf = kf + (size_t)NNZ * NKV * HD;            // [NNZ, NKV*HD]  8 MB
  half_t* af = vf + (size_t)NNZ * NKV * HD;            // [NNZ, NH*HD]  32 MB

  dim3 blk(256);
  // QKV projections (A=f32 hidden, out=f16)
  gemm_wmma<false, false><<<dim3(HID / 128, NNZ / 64), blk, 0, stream>>>(
      hidden, wq, qf, HID, HID);
  gemm_wmma<false, false><<<dim3((NKV * HD) / 128, NNZ / 64), blk, 0, stream>>>(
      hidden, wk, kf, NKV * HD, HID);
  gemm_wmma<false, false><<<dim3((NKV * HD) / 128, NNZ / 64), blk, 0, stream>>>(
      hidden, wv, vf, NKV * HD, HID);

  // RoPE on q and k
  rope_kernel<<<(NNZ * NH * 64 + 255) / 256, blk, 0, stream>>>(qf, pos, NH);
  rope_kernel<<<(NNZ * NKV * 64 + 255) / 256, blk, 0, stream>>>(kf, pos, NKV);

  // causal GQA flash attention
  attn_kernel<<<BATCH * NH * (SEQ / 128), blk, 0, stream>>>(qf, kf, vf, af);

  // output projection (A=f16 attn via double-buffered TDM, out=f32)
  gemm_wmma<true, true><<<dim3(HID / 128, NNZ / 64), blk, 0, stream>>>(
      af, wo, d_out, HID, HID);
}
